// ContrastiveLoss_8684423872787
// MI455X (gfx1250) — matrix operand order repaired and checked by reading, not compiled, not run on hardware
//
#include <hip/hip_runtime.h>
#include <math.h>

// Problem constants (match reference)
#define N_  256   // images == captions
#define R_  36    // regions
#define L_  50    // max words
#define D_  256   // feature dim
#define RP  48    // regions padded to multiple of 16 (WMMA M)
#define LP  64    // words padded (WMMA N)
#define KP  64    // padded K for Gram GEMM (regions padded to mult of 32)

#define LAMBDA_SOFTMAX 9.0f
#define LAMBDA_LSE     6.0f
#define MARGIN         0.2f

typedef __attribute__((ext_vector_type(16))) __bf16 v16bf;
typedef __attribute__((ext_vector_type(8)))  float  v8f;

union Frag { uint4 u4[2]; v16bf v; };

static __device__ inline unsigned short f2bf(float f) {
    unsigned int u = __float_as_uint(f);
    unsigned int r = (u + 0x7fffu + ((u >> 16) & 1u)) >> 16;   // RNE
    return (unsigned short)r;
}
static __device__ inline float bf2f(unsigned short h) {
    return __uint_as_float(((unsigned int)h) << 16);
}

static __device__ inline v8f wmma_bf16(const Frag& a, const Frag& b, v8f c) {
    return __builtin_amdgcn_wmma_f32_16x16x32_bf16(
        /*neg_a=*/false, a.v, /*neg_b=*/false, b.v,
        /*c_mod=*/(short)0, c, /*reuse_a=*/false, /*reuse_b=*/false);
}

// A fragment: 16x32 bf16, row m, K-window kk; per documented layout:
// lane<16: elems 0..7 -> K=kk+0..7,  elems 8..15 -> K=kk+16..23
// lane>=16: elems 0..7 -> K=kk+8..15, elems 8..15 -> K=kk+24..31
static __device__ inline Frag loadA(const unsigned short* rowM, int kk, int hi) {
    Frag f;
    f.u4[0] = *(const uint4*)(rowM + kk + hi * 8);
    f.u4[1] = *(const uint4*)(rowM + kk + 16 + hi * 8);
    return f;
}
// B fragment: 32x16 bf16 from row-major [n][k] storage (i.e. B-transposed rows):
// same K order as A: lane<16: K=kk+0..7, kk+16..23; lane>=16: K=kk+8..15, kk+24..31
static __device__ inline Frag loadB(const unsigned short* rowN, int kk, int hi) {
    Frag f;
    const unsigned short* p = rowN + kk + hi * 8;
    f.u4[0] = *(const uint4*)(p);
    f.u4[1] = *(const uint4*)(p + 16);
    return f;
}

// ---------------------------------------------------------------------------
// Kernel 1: fp32 -> bf16 (zero-padded) copies of im / captions + caption word norms
// ---------------------------------------------------------------------------
__global__ __launch_bounds__(256) void prep_kernel(
    const float* __restrict__ im, const float* __restrict__ s,
    unsigned short* __restrict__ imb,   // [N][RP][D]
    unsigned short* __restrict__ capb,  // [N][LP][D]
    float* __restrict__ cn)             // [N][LP]
{
    size_t stride = (size_t)gridDim.x * blockDim.x;
    size_t t0 = (size_t)blockIdx.x * blockDim.x + threadIdx.x;

    const size_t nIm = (size_t)N_ * RP * D_;
    for (size_t idx = t0; idx < nIm; idx += stride) {
        int d = (int)(idx & (D_ - 1));
        int r = (int)((idx / D_) % RP);
        int i = (int)(idx / ((size_t)RP * D_));
        float v = (r < R_) ? im[((size_t)i * R_ + r) * D_ + d] : 0.f;
        imb[idx] = f2bf(v);
    }
    const size_t nCap = (size_t)N_ * LP * D_;
    for (size_t idx = t0; idx < nCap; idx += stride) {
        int d = (int)(idx & (D_ - 1));
        int l = (int)((idx / D_) % LP);
        int j = (int)(idx / ((size_t)LP * D_));
        float v = (l < L_) ? s[((size_t)j * L_ + l) * D_ + d] : 0.f;
        capb[idx] = f2bf(v);
    }
    const size_t nCn = (size_t)N_ * LP;
    for (size_t idx = t0; idx < nCn; idx += stride) {
        int l = (int)(idx & (LP - 1));
        int j = (int)(idx / LP);
        float ss = 0.f;
        if (l < L_) {
            const float* p = s + ((size_t)j * L_ + l) * D_;
            for (int d = 0; d < D_; d++) { float v = p[d]; ss += v * v; }
        }
        cn[idx] = sqrtf(ss);
    }
}

// ---------------------------------------------------------------------------
// Kernel 2: per-image Gram matrices G_i = im_i @ im_i^T (bf16, padded [RP][KP])
// One block (4 waves) per image; 48x48x256 GEMM via WMMA. B fragment comes
// straight from row-major im (G symmetric => B^T rows == im rows).
// ---------------------------------------------------------------------------
__global__ __launch_bounds__(128) void gram_kernel(
    const unsigned short* __restrict__ imb, unsigned short* __restrict__ gramb)
{
    int i = blockIdx.x;
    int lane = threadIdx.x & 31, wave = threadIdx.x >> 5;
    int l15 = lane & 15, hi = lane >> 4;
    const unsigned short* base = imb + (size_t)i * RP * D_;
    unsigned short* g = gramb + (size_t)i * RP * KP;

    for (int t = wave; t < 9; t += 4) {          // uniform per wave: EXEC all-1s
        int mt = t / 3, nt = t % 3;
        v8f acc = {};
        const unsigned short* aRow = base + (size_t)(mt * 16 + l15) * D_;
        const unsigned short* bRow = base + (size_t)(nt * 16 + l15) * D_;
        for (int kk = 0; kk < D_; kk += 32) {
            Frag a = loadA(aRow, kk, hi);
            Frag b = loadB(bRow, kk, hi);
            acc = wmma_bf16(a, b, acc);
        }
        for (int c = 0; c < 8; c++) {
            int m = mt * 16 + c + hi * 8;
            int n = nt * 16 + l15;
            g[m * KP + n] = f2bf(acc[c]);
        }
    }
    // zero-pad K columns 48..63 (rows 36..47 are already zero via padded im)
    for (int idx = threadIdx.x; idx < RP * 16; idx += 128) {
        int m = idx / 16, n = RP + (idx & 15);
        g[m * KP + n] = 0;
    }
}

// ---------------------------------------------------------------------------
// Kernel 3: one block per (image i, caption j). Produces scores[i][j].
//   GEMM1 (WMMA): raw[48][64] = im_i(48x256) @ cap_j^T(256x64)
//   elementwise:  leaky+l2norm over words, softmax over regions -> b (bf16)
//   num[l] = sum_r b[l][r]*raw[r][l]   (reuses raw, no w needed)
//   GEMM2 (WMMA): t[64][48] = b_pad(64x64) @ G_i(64x48);  wn2[l]=sum_r t*b
//   sim -> masked LSE -> score
// ---------------------------------------------------------------------------
__global__ __launch_bounds__(128) void xattn_score_kernel(
    const unsigned short* __restrict__ imb,    // [N][RP][D]
    const unsigned short* __restrict__ capb,   // [N][LP][D]
    const unsigned short* __restrict__ gramb,  // [N][RP][KP]
    const float* __restrict__ cn,              // [N][LP]
    const int* __restrict__ cap_lens,          // [N]
    float* __restrict__ scores)                // [N_img][N_cap]
{
    int i = blockIdx.x;   // image
    int j = blockIdx.y;   // caption

    __shared__ float raw[RP * LP];                                  // 12 KB [r][l]
    __shared__ float tbuf[LP * RP];                                 // 12 KB (attn [r][l], then t [l][r'])
    __shared__ unsigned short bTbf[LP * KP] __attribute__((aligned(16))); // 8 KB [l][r-pad]
    __shared__ float numv[LP];
    __shared__ float zv[LP];

    int tid = threadIdx.x;
    int lane = tid & 31, wave = tid >> 5;
    int l15 = lane & 15, hi = lane >> 4;
    int len = cap_lens[j];

    const unsigned short* imBase  = imb  + (size_t)i * RP * D_;
    const unsigned short* capBase = capb + (size_t)j * LP * D_;

    // ---- GEMM1: wave w owns word-tile w (N), all 3 region-tiles (M) ----
    v8f acc0 = {}, acc1 = {}, acc2 = {};
    {
        const unsigned short* bRow = capBase + (size_t)(wave * 16 + l15) * D_;
        const unsigned short* aRow0 = imBase + (size_t)(0  + l15) * D_;
        const unsigned short* aRow1 = imBase + (size_t)(16 + l15) * D_;
        const unsigned short* aRow2 = imBase + (size_t)(32 + l15) * D_;
        for (int kk = 0; kk < D_; kk += 32) {
            Frag b = loadB(bRow, kk, hi);
            Frag a0 = loadA(aRow0, kk, hi);
            acc0 = wmma_bf16(a0, b, acc0);
            Frag a1 = loadA(aRow1, kk, hi);
            acc1 = wmma_bf16(a1, b, acc1);
            Frag a2 = loadA(aRow2, kk, hi);
            acc2 = wmma_bf16(a2, b, acc2);
        }
    }
    {
        int n = wave * 16 + l15;
        for (int c = 0; c < 8; c++) {
            raw[(0  + c + hi * 8) * LP + n] = acc0[c];
            raw[(16 + c + hi * 8) * LP + n] = acc1[c];
            raw[(32 + c + hi * 8) * LP + n] = acc2[c];
        }
    }
    __syncthreads();

    // ---- leaky_relu + mask + l2norm over words, fold lambda_softmax ----
    if (tid < R_) {
        int r = tid;
        float ss = 0.f;
        for (int l = 0; l < len; l++) {
            float v = raw[r * LP + l];
            v = (v > 0.f) ? v : 0.1f * v;
            ss += v * v;
        }
        float sc = LAMBDA_SOFTMAX / (sqrtf(ss) + 1e-8f);
        for (int l = 0; l < LP; l++) {
            float v = raw[r * LP + l];
            v = (v > 0.f) ? v : 0.1f * v;
            tbuf[r * LP + l] = (l < len) ? v * sc : 0.f;
        }
    }
    __syncthreads();

    // ---- softmax over regions per word column; num[l] = sum_r b*raw ----
    if (tid < LP) {
        int l = tid;
        float mx = -1e30f;
        for (int r = 0; r < R_; r++) mx = fmaxf(mx, tbuf[r * LP + l]);
        float ssum = 0.f;
        for (int r = 0; r < R_; r++) {
            float e = __expf(tbuf[r * LP + l] - mx);
            tbuf[r * LP + l] = e;
            ssum += e;
        }
        float inv = 1.f / ssum;
        float nm = 0.f;
        for (int r = 0; r < R_; r++) {
            float b = tbuf[r * LP + l] * inv;
            nm += b * raw[r * LP + l];
            bTbf[l * KP + r] = f2bf(b);
        }
        for (int r = R_; r < KP; r++) bTbf[l * KP + r] = 0;
        numv[l] = nm;
    }
    __syncthreads();

    // ---- GEMM2: t = b_pad(64x64) @ G_i(64x48); wave w owns word-tile w ----
    {
        v8f t0 = {}, t1 = {}, t2 = {};
        const unsigned short* gBase = gramb + (size_t)i * RP * KP;
        const unsigned short* aRow = &bTbf[(wave * 16 + l15) * KP];
        const unsigned short* bRow0 = gBase + (size_t)(0  + l15) * KP;
        const unsigned short* bRow1 = gBase + (size_t)(16 + l15) * KP;
        const unsigned short* bRow2 = gBase + (size_t)(32 + l15) * KP;
        for (int kk = 0; kk < KP; kk += 32) {
            Frag a = loadA(aRow, kk, hi);
            Frag b0 = loadB(bRow0, kk, hi);
            t0 = wmma_bf16(a, b0, t0);
            Frag b1 = loadB(bRow1, kk, hi);
            t1 = wmma_bf16(a, b1, t1);
            Frag b2 = loadB(bRow2, kk, hi);
            t2 = wmma_bf16(a, b2, t2);
        }
        for (int c = 0; c < 8; c++) {
            int l0 = wave * 16 + c + hi * 8;
            tbuf[l0 * RP + (0  + l15)] = t0[c];
            tbuf[l0 * RP + (16 + l15)] = t1[c];
            tbuf[l0 * RP + (32 + l15)] = t2[c];
        }
    }
    __syncthreads();

    // ---- cosine sim + masked z ----
    if (tid < LP) {
        int l = tid;
        float w2 = 0.f;
        for (int r = 0; r < R_; r++)
            w2 += tbuf[l * RP + r] * bf2f(bTbf[l * KP + r]);
        float c = cn[(size_t)j * LP + l];
        float sim = numv[l] / fmaxf(sqrtf(fmaxf(w2, 0.f)) * c, 1e-8f);
        zv[l] = (l < len) ? sim * LAMBDA_LSE : -1e30f;
    }
    __syncthreads();

    // ---- LSE over valid words ----
    if (tid == 0) {
        float mx = -1e30f;
        for (int l = 0; l < len; l++) mx = fmaxf(mx, zv[l]);
        float ssum = 0.f;
        for (int l = 0; l < len; l++) ssum += __expf(zv[l] - mx);
        scores[(size_t)i * N_ + j] = (logf(ssum) + mx) / LAMBDA_LSE;
    }
}

// ---------------------------------------------------------------------------
// Kernel 4: contrastive hinge loss reduction over the 256x256 score matrix
// ---------------------------------------------------------------------------
__global__ __launch_bounds__(256) void loss_kernel(
    const float* __restrict__ scores, float* __restrict__ out)
{
    __shared__ float red[256];
    float acc = 0.f;
    for (int idx = threadIdx.x; idx < N_ * N_; idx += 256) {
        int i = idx >> 8, j = idx & 255;
        if (i != j) {
            float sij = scores[idx];
            float di = scores[i * (N_ + 1)];
            float dj = scores[j * (N_ + 1)];
            acc += fmaxf(MARGIN + sij - di, 0.f) + fmaxf(MARGIN + sij - dj, 0.f);
        }
    }
    red[threadIdx.x] = acc;
    __syncthreads();
    for (int s = 128; s > 0; s >>= 1) {
        if (threadIdx.x < s) red[threadIdx.x] += red[threadIdx.x + s];
        __syncthreads();
    }
    if (threadIdx.x == 0) out[0] = red[0];
}

// ---------------------------------------------------------------------------
extern "C" void kernel_launch(void* const* d_in, const int* in_sizes, int n_in,
                              void* d_out, int out_size, void* d_ws, size_t ws_size,
                              hipStream_t stream)
{
    (void)in_sizes; (void)n_in; (void)out_size; (void)ws_size;
    const float* im = (const float*)d_in[0];        // (N, R, D) fp32
    const float* s  = (const float*)d_in[1];        // (N, L, D) fp32
    const int* cap_lens = (const int*)d_in[2];      // (N,) int32

    // Workspace carve-up (~16.6 MB total, all offsets 256B-aligned)
    char* ws = (char*)d_ws;
    unsigned short* imb   = (unsigned short*)ws; ws += (size_t)N_ * RP * D_ * 2; // 6.29 MB
    unsigned short* capb  = (unsigned short*)ws; ws += (size_t)N_ * LP * D_ * 2; // 8.39 MB
    unsigned short* gramb = (unsigned short*)ws; ws += (size_t)N_ * RP * KP * 2; // 1.57 MB
    float* cn     = (float*)ws; ws += (size_t)N_ * LP * 4;                       // 64 KB
    float* scores = (float*)ws;                                                  // 256 KB

    prep_kernel<<<1024, 256, 0, stream>>>(im, s, imb, capb, cn);
    gram_kernel<<<N_, 128, 0, stream>>>(imb, gramb);
    dim3 grid(N_, N_);
    xattn_score_kernel<<<grid, 128, 0, stream>>>(imb, capb, gramb, cn, cap_lens, scores);
    loss_kernel<<<1, 256, 0, stream>>>(scores, (float*)d_out);
}
